// PropertyFocusedAttention_6021544149503
// MI455X (gfx1250) — matrix operand order repaired and checked by reading, not compiled, not run on hardware
//
#include <hip/hip_runtime.h>

#define BATCH  16384
#define SEQ    5
#define DMODEL 512
#define NHEAD  8
#define HDIM   64
#define BPW    16                 // batches per workgroup
#define ROWS   (BPW * SEQ)        // 80 rows = 5 M-tiles of 16
#define PITCH  520                // LDS row pitch (bf16 elems) -> breaks bank conflicts
#define KTILES (DMODEL / 32)      // 16 K-slices of 32 per WMMA
#define MTILES (ROWS / 16)        // 5
#define STEPS  (MTILES * KTILES)  // 80 flattened (mt,kt) steps

typedef __attribute__((ext_vector_type(16))) __bf16 v16bf;
typedef __attribute__((ext_vector_type(8)))  float  v8f;

union FragU { uint4 u[2]; v16bf v; };

__device__ __forceinline__ unsigned short f2bf(float f) {
  unsigned int u = __float_as_uint(f);
  u += 0x7fffu + ((u >> 16) & 1u);          // round-to-nearest-even
  return (unsigned short)(u >> 16);
}
__device__ __forceinline__ float bf2f(unsigned short h) {
  return __uint_as_float(((unsigned int)h) << 16);
}

// ---------------------------------------------------------------------------
// Pre-pass: convert W (f32, [k][n]) -> bf16 transposed [n][k] so WMMA
// B-fragments are contiguous b128 loads. 4 weights -> 2 MB in d_ws (L2-resident).
// ---------------------------------------------------------------------------
extern "C" __global__ __launch_bounds__(256) void cvt_wt_kernel(
    const float* __restrict__ Wq, const float* __restrict__ Wk,
    const float* __restrict__ Wv, const float* __restrict__ Wo,
    unsigned short* __restrict__ wt)
{
  int idx  = blockIdx.x * 256 + threadIdx.x;      // 4 * 512 * 512 elements
  int proj = idx >> 18;
  int r    = idx & ((1 << 18) - 1);
  int n    = r >> 9;
  int k    = r & 511;
  const float* W = (proj == 0) ? Wq : (proj == 1) ? Wk : (proj == 2) ? Wv : Wo;
  wt[idx] = f2bf(W[k * DMODEL + n]);
}

// ---------------------------------------------------------------------------
// 80x512 row-block times 512x512^T, bf16 WMMA, f32 accumulate.
// Each wave owns 4 N-tiles. Per N-tile: B fragments for the FULL K dimension
// are register-resident (16 frags = 128 VGPRs, weights read once per wave),
// then all 5 M-tiles x 16 K-slices are swept as one flattened, fully unrolled
// 80-step stream with a 4-deep rotating A-fragment pipeline: step i issues the
// ds_load_b128 pair for step i+3 and consumes buffer i&3, keeping 3 fragment
// loads in flight ahead of every WMMA (waits become dscnt<=6, not 0).
// ---------------------------------------------------------------------------
__device__ __forceinline__ void gemm80x512(
    const unsigned short* sArows,              // LDS A [ROWS][PITCH] bf16
    const unsigned short* __restrict__ wtN,    // global B^T [512][512] bf16 (n-major)
    const float* __restrict__ bias,            // [512]
    unsigned short* sOut,                      // LDS out [ROWS][PITCH] (or nullptr)
    float* __restrict__ gOut,                  // global out, row stride DMODEL (or nullptr)
    int wave, int lane)
{
  const int half = lane >> 4;
  const int l16  = lane & 15;
  const unsigned short* abase = sArows + l16 * PITCH + half * 8;

#pragma unroll 1
  for (int nt = wave * 4; nt < wave * 4 + 4; ++nt) {
    const int n = nt * 16 + l16;
    const unsigned short* brow = wtN + n * DMODEL;

    // Hoist all B fragments for this column tile: resident across the M sweep.
    FragU b[KTILES];
#pragma unroll
    for (int kt = 0; kt < KTILES; ++kt) {
      // B 32x16 bf16 fragment: lane = column n; u[0] at K half*8, u[1] at K 16 + half*8
      b[kt].u[0] = *(const uint4*)(brow + kt * 32 + half * 8);
      b[kt].u[1] = *(const uint4*)(brow + kt * 32 + half * 8 + 16);
    }
    const float bb = bias[n];

    // A 16x32 bf16 fragment for step idx (mt = idx/16, kt = idx%16):
    //   u[0] at row(mt)+l16, K kb + half*8 ; u[1] at K kb + 16 + half*8
    FragU ab[4];
#define ALOAD(BUF, IDX)                                                        \
    {                                                                          \
      const unsigned short* _p = abase + ((IDX) >> 4) * (16 * PITCH)           \
                               + ((IDX) & 15) * 32;                            \
      ab[BUF].u[0] = *(const uint4*)(_p);                                      \
      ab[BUF].u[1] = *(const uint4*)(_p + 16);                                 \
    }

    ALOAD(0, 0) ALOAD(1, 1) ALOAD(2, 2)      // prologue: 3 steps in flight
    v8f acc = {};
#pragma unroll
    for (int idx = 0; idx < STEPS; ++idx) {
      if (idx + 3 < STEPS) ALOAD((idx + 3) & 3, idx + 3)
      acc = __builtin_amdgcn_wmma_f32_16x16x32_bf16(false, ab[idx & 3].v, false,
                                                    b[idx & 15].v, (short)0, acc,
                                                    false, false);
      if ((idx & 15) == 15) {                // finished one 16x16 output tile
        const int mt    = idx >> 4;
        const int rbase = mt * 16 + half * 8;  // C layout: VGPR r -> row half*8+r
        if (sOut) {
#pragma unroll
          for (int r = 0; r < 8; ++r)
            sOut[(rbase + r) * PITCH + n] = f2bf(acc[r] + bb);
        } else {
#pragma unroll
          for (int r = 0; r < 8; ++r)
            gOut[(size_t)(rbase + r) * DMODEL + n] = acc[r] + bb;
        }
        acc = v8f{};
      }
    }
#undef ALOAD
  }
}

// ---------------------------------------------------------------------------
// Fused attention: one workgroup = 16 batches (80 rows). Everything between
// reading x and writing out stays in LDS.
// ---------------------------------------------------------------------------
extern "C" __global__ __launch_bounds__(256) void fused_attn_kernel(
    const float* __restrict__ x,
    const unsigned short* __restrict__ wt,     // [4][512][512] bf16 transposed
    const float* __restrict__ bq, const float* __restrict__ bk,
    const float* __restrict__ bv, const float* __restrict__ bo,
    const float* __restrict__ prop_bias,       // [H][S][S] f32
    float* __restrict__ out)                   // [B][S][D] f32
{
  extern __shared__ unsigned short smem[];
  unsigned short* sX = smem;                         // x block (later dead)
  unsigned short* sA = sX + ROWS * PITCH;            // Q, then V
  unsigned short* sB = sA + ROWS * PITCH;            // K, then O
  float*          sP = (float*)(sB + ROWS * PITCH);  // probs [BPW][H][S][8]

  const int tid  = threadIdx.x;
  const int lane = tid & 31;
  const int wave = tid >> 5;
  const size_t rowBase = (size_t)blockIdx.x * ROWS;  // first global token row

  // Phase 0: x (f32) -> LDS bf16
  for (int i = tid; i < ROWS * (DMODEL / 4); i += 256) {
    const int r = i >> 7, c4 = i & 127;
    const float4 vv = *(const float4*)(x + (rowBase + r) * DMODEL + c4 * 4);
    const unsigned int lo = (unsigned int)f2bf(vv.x) | ((unsigned int)f2bf(vv.y) << 16);
    const unsigned int hi = (unsigned int)f2bf(vv.z) | ((unsigned int)f2bf(vv.w) << 16);
    *(uint2*)(sX + r * PITCH + c4 * 4) = make_uint2(lo, hi);
  }
  __syncthreads();

  // Phase 1: Q = X Wq^T + bq  -> sA
  gemm80x512(sX, wt + 0 * DMODEL * DMODEL, bq, sA, nullptr, wave, lane);
  __syncthreads();
  // Phase 2: K = X Wk^T + bk  -> sB
  gemm80x512(sX, wt + 1 * DMODEL * DMODEL, bk, sB, nullptr, wave, lane);
  __syncthreads();

  // Phase 3: 5x5 scores + prop_bias + softmax. One thread per (batch, head).
  if (tid < BPW * NHEAD) {
    const int b = tid >> 3, h = tid & 7;
    const unsigned short* qb = sA + (b * SEQ) * PITCH + h * HDIM;
    const unsigned short* kb = sB + (b * SEQ) * PITCH + h * HDIM;
    float sc[SEQ][SEQ];
#pragma unroll
    for (int s = 0; s < SEQ; ++s) {
#pragma unroll
      for (int t = 0; t < SEQ; ++t) {
        float d = 0.f;
        for (int e = 0; e < HDIM; ++e)
          d += bf2f(qb[s * PITCH + e]) * bf2f(kb[t * PITCH + e]);
        sc[s][t] = d * 0.125f + prop_bias[(h * SEQ + s) * SEQ + t];  // 1/sqrt(64)
      }
    }
    float* pOut = sP + tid * SEQ * 8;
#pragma unroll
    for (int s = 0; s < SEQ; ++s) {
      float mx = sc[s][0];
      for (int t = 1; t < SEQ; ++t) mx = fmaxf(mx, sc[s][t]);
      float e_[SEQ], sum = 0.f;
      for (int t = 0; t < SEQ; ++t) { e_[t] = __expf(sc[s][t] - mx); sum += e_[t]; }
      const float inv = 1.f / sum;
      for (int t = 0; t < SEQ; ++t) pOut[s * 8 + t] = e_[t] * inv;
    }
  }
  __syncthreads();

  // Phase 4: V = X Wv^T + bv  -> sA (Q is dead)
  gemm80x512(sX, wt + 2 * DMODEL * DMODEL, bv, sA, nullptr, wave, lane);
  __syncthreads();

  // Phase 5: O = P * V  -> sB (K is dead). Dot-of-5 per element, f32 VALU.
  for (int i = tid; i < ROWS * DMODEL; i += 256) {
    const int row = i >> 9, col = i & 511;
    const int b = row / SEQ, s = row % SEQ, h = col >> 6;
    const float* p = sP + ((b * NHEAD + h) * SEQ + s) * 8;
    float acc = 0.f;
#pragma unroll
    for (int t = 0; t < SEQ; ++t)
      acc += p[t] * bf2f(sA[(b * SEQ + t) * PITCH + col]);
    sB[row * PITCH + col] = f2bf(acc);
  }
  __syncthreads();

  // Phase 6: out = O Wo^T + bo  -> global f32
  gemm80x512(sB, wt + 3 * DMODEL * DMODEL, bo, nullptr,
             out + rowBase * DMODEL, wave, lane);
}

// ---------------------------------------------------------------------------
extern "C" void kernel_launch(void* const* d_in, const int* in_sizes, int n_in,
                              void* d_out, int out_size, void* d_ws, size_t ws_size,
                              hipStream_t stream)
{
  const float* x  = (const float*)d_in[0];
  const float* Wq = (const float*)d_in[1];
  const float* bq = (const float*)d_in[2];
  const float* Wk = (const float*)d_in[3];
  const float* bk = (const float*)d_in[4];
  const float* Wv = (const float*)d_in[5];
  const float* bv = (const float*)d_in[6];
  const float* Wo = (const float*)d_in[7];
  const float* bo = (const float*)d_in[8];
  const float* pb = (const float*)d_in[9];
  unsigned short* wt = (unsigned short*)d_ws;   // 4*512*512 bf16 = 2 MB

  cvt_wt_kernel<<<dim3(4 * DMODEL * DMODEL / 256), dim3(256), 0, stream>>>(
      Wq, Wk, Wv, Wo, wt);

  const size_t smem = (size_t)3 * ROWS * PITCH * sizeof(unsigned short)
                    + (size_t)BPW * NHEAD * SEQ * 8 * sizeof(float);   // 270,080 B
  fused_attn_kernel<<<dim3(BATCH / BPW), dim3(256), smem, stream>>>(
      x, wt, bq, bk, bv, bo, pb, (float*)d_out);
}
